// AgentAttention_38878043963493
// MI455X (gfx1250) — compile-verified, matrix operand order checked by reading, not run on hardware
//
#include <hip/hip_runtime.h>
#include <hip/hip_bf16.h>
#include <stddef.h>

// ---------------------------------------------------------------------------
// Types for CDNA5 WMMA (gfx1250, wave32)
// ---------------------------------------------------------------------------
typedef __attribute__((ext_vector_type(16))) __bf16 v16bf;
typedef __attribute__((ext_vector_type(8)))  float  v8f;
typedef __attribute__((ext_vector_type(4)))  int    v4i;

union Frag16 {
    uint4 q[2];   // 32 bytes
    v16bf v;      // 16 x bf16 = 32 bytes
};

__device__ __forceinline__ unsigned int bfbits(float f) {
    union { float f; unsigned int u; } x;
    x.f = f;
    unsigned int u = x.u;
    return (u + 0x7FFFu + ((u >> 16) & 1u)) >> 16;   // round-to-nearest-even
}

// ---------------------------------------------------------------------------
// Async global->LDS staging (gfx1250). Guarded: falls back to sync b128 copy.
// Builtin signature (from clang-22 diagnostics): param 1 = AS1 (global) v4i*,
// param 2 = AS3 (LDS) v4i*, then imm offset and imm cpol.
// ---------------------------------------------------------------------------
#if defined(__has_builtin)
#  if __has_builtin(__builtin_amdgcn_global_load_async_to_lds_b128)
#    define USE_ASYNC_LDS 1
#  endif
#endif
#ifndef USE_ASYNC_LDS
#  define USE_ASYNC_LDS 0
#endif

#if USE_ASYNC_LDS
__device__ __forceinline__ void async_cp16(const unsigned short* g, unsigned short* l) {
    __builtin_amdgcn_global_load_async_to_lds_b128(
        (__attribute__((address_space(1))) v4i*)g,
        (__attribute__((address_space(3))) v4i*)l, 0, 0);
}
__device__ __forceinline__ void async_wait0() {
#  if __has_builtin(__builtin_amdgcn_s_wait_asynccnt)
    __builtin_amdgcn_s_wait_asynccnt(0);
#  else
    asm volatile("s_wait_asynccnt 0" ::: "memory");
#  endif
}
#endif

// ---------------------------------------------------------------------------
// Tiled GEMM:  C[M,N] = A[M,K] * BT[N,K]^T (+ bias[N])
// A: bf16 row-major [M][K].  BT: bf16 N-major [N][K] (pre-transposed weight).
// f32 accumulate via v_wmma_f32_16x16x32_bf16.
// Block tile 128x128, BK=32, 256 threads = 8 wave32s, LDS double-buffered,
// tiles staged with GLOBAL_LOAD_ASYNC_TO_LDS_B128 (ASYNCcnt) when available.
// Wave w: 32-row strip (w>>1)*32 (2 m-tiles), column half (w&1)*64 (4 n-tiles)
//   -> 8 WMMA per wave per k-step; B fragments reused across 2 m-tiles.
// Requires M%128==0, N%128==0, K%32==0.
// ---------------------------------------------------------------------------
#define GEMM_BM 128
#define GEMM_BN 128
#define GEMM_BK 32
#define LDS_STRIDE 40   // halfs; 80B row pitch (16B aligned, bank-spread)

__global__ __launch_bounds__(256) void gemm_bf16_wmma(
    const unsigned short* __restrict__ A, const unsigned short* __restrict__ BT,
    const float* __restrict__ bias, float* __restrict__ C,
    int M, int N, int K)
{
    __shared__ alignas(16) unsigned short As[2][GEMM_BM][LDS_STRIDE];
    __shared__ alignas(16) unsigned short Bs[2][GEMM_BN][LDS_STRIDE];

    const int tid     = threadIdx.x;
    const int wave    = tid >> 5;
    const int lane    = tid & 31;
    const int rowBase = blockIdx.y * GEMM_BM;
    const int colBase = blockIdx.x * GEMM_BN;

    const int rStrip = wave >> 1;   // 0..3 -> 32-row strip
    const int cHalf  = wave & 1;    // 0..1 -> 64-col half

    v8f acc[2][4];
#pragma unroll
    for (int s = 0; s < 2; ++s)
#pragma unroll
        for (int j = 0; j < 4; ++j) acc[s][j] = {};

    const int laneN = lane & 15;
    const int kBase = (lane < 16) ? 0 : 8;   // ISA 16-bit A/B lane K-phase
    const int aRow0 = (rStrip << 5) + laneN;
    const int aRow1 = aRow0 + 16;

    // staging: 512 granules of 16B per tile (128 rows x 4 granules), 2/thread
    const int gr0 = tid >> 2;            // row 0..63
    const int gr1 = gr0 + 64;            // row 64..127
    const int gc  = (tid & 3) << 3;      // half-offset within row: 0,8,16,24

    const unsigned short* aG0 = A  + (size_t)(rowBase + gr0) * K + gc;
    const unsigned short* aG1 = A  + (size_t)(rowBase + gr1) * K + gc;
    const unsigned short* bG0 = BT + (size_t)(colBase + gr0) * K + gc;
    const unsigned short* bG1 = BT + (size_t)(colBase + gr1) * K + gc;

    const int nIter = K / GEMM_BK;

#if USE_ASYNC_LDS
#define STAGE_TILE(BUF, K0)                                                  \
    do {                                                                     \
        async_cp16(aG0 + (K0), &As[BUF][gr0][gc]);                           \
        async_cp16(aG1 + (K0), &As[BUF][gr1][gc]);                           \
        async_cp16(bG0 + (K0), &Bs[BUF][gr0][gc]);                           \
        async_cp16(bG1 + (K0), &Bs[BUF][gr1][gc]);                           \
    } while (0)

    STAGE_TILE(0, 0);
    async_wait0();
    __syncthreads();

    for (int kt = 0; kt < nIter; ++kt) {
        const int cur = kt & 1;
        const bool more = (kt + 1) < nIter;
        if (more) STAGE_TILE(1 - cur, (kt + 1) * GEMM_BK);
#else
    uint4 sA0, sA1, sB0, sB1;
#define LOAD_TILE(K0)                                                        \
    do {                                                                     \
        sA0 = *(const uint4*)(aG0 + (K0));                                   \
        sA1 = *(const uint4*)(aG1 + (K0));                                   \
        sB0 = *(const uint4*)(bG0 + (K0));                                   \
        sB1 = *(const uint4*)(bG1 + (K0));                                   \
    } while (0)
#define STORE_TILE(BUF)                                                      \
    do {                                                                     \
        *(uint4*)&As[BUF][gr0][gc] = sA0;                                    \
        *(uint4*)&As[BUF][gr1][gc] = sA1;                                    \
        *(uint4*)&Bs[BUF][gr0][gc] = sB0;                                    \
        *(uint4*)&Bs[BUF][gr1][gc] = sB1;                                    \
    } while (0)

    LOAD_TILE(0);
    STORE_TILE(0);
    __syncthreads();

    for (int kt = 0; kt < nIter; ++kt) {
        const int cur = kt & 1;
        const bool more = (kt + 1) < nIter;
        if (more) LOAD_TILE((kt + 1) * GEMM_BK);
#endif

        // ---- fragments per ISA 16-bit layout:
        // lane<16: K 0..7 (elems 0..7), K 16..23 (elems 8..15)
        // lane>=16: K 8..15, K 24..31
        Frag16 fa0, fa1;
        fa0.q[0] = *(const uint4*)&As[cur][aRow0][kBase];
        fa0.q[1] = *(const uint4*)&As[cur][aRow0][kBase + 16];
        fa1.q[0] = *(const uint4*)&As[cur][aRow1][kBase];
        fa1.q[1] = *(const uint4*)&As[cur][aRow1][kBase + 16];

#pragma unroll
        for (int j = 0; j < 4; ++j) {
            const int bCol = (cHalf << 6) + (j << 4) + laneN;
            Frag16 fb;
            fb.q[0] = *(const uint4*)&Bs[cur][bCol][kBase];
            fb.q[1] = *(const uint4*)&Bs[cur][bCol][kBase + 16];
            acc[0][j] = __builtin_amdgcn_wmma_f32_16x16x32_bf16(
                false, fa0.v, false, fb.v, (short)0, acc[0][j], false, false);
            acc[1][j] = __builtin_amdgcn_wmma_f32_16x16x32_bf16(
                false, fa1.v, false, fb.v, (short)0, acc[1][j], false, false);
        }

#if USE_ASYNC_LDS
        if (more) async_wait0();
        __syncthreads();
    }
#undef STAGE_TILE
#else
        if (more) STORE_TILE(1 - cur);
        __syncthreads();
    }
#undef LOAD_TILE
#undef STORE_TILE
#endif

    // ---- write back: C/D layout: lane<16 -> M=v, lane>=16 -> M=v+8, N=lane&15
#pragma unroll
    for (int s = 0; s < 2; ++s) {
        const int cRowBase = rowBase + (rStrip << 5) + (s << 4) + ((lane < 16) ? 0 : 8);
#pragma unroll
        for (int j = 0; j < 4; ++j) {
            const int col = colBase + (cHalf << 6) + (j << 4) + laneN;
            const float bv = bias ? bias[col] : 0.0f;
#pragma unroll
            for (int v = 0; v < 8; ++v) {
                const int row = cRowBase + v;
                C[(size_t)row * N + col] = acc[s][j][v] + bv;
            }
        }
    }
}

// ---------------------------------------------------------------------------
// Problem constants
// ---------------------------------------------------------------------------
#define AB 4
#define AT 3584
#define AH 16
#define AD 64
#define AP 7
#define ADIM 1024
#define QKVC 3072
#define TP 512          // T / P
#define SCALE 0.125f    // 64^-0.5

// qkv buffer layout: [B*T, 3072]; q at c = h*64+d, k at 1024+h*64+d, v at 2048+h*64+d

// ---------------------------------------------------------------------------
// f32 -> bf16 elementwise convert (4 per thread). n % 1024 == 0.
// ---------------------------------------------------------------------------
__global__ __launch_bounds__(256) void cvt_f32_bf16(
    const float* __restrict__ in, unsigned short* __restrict__ out, size_t n)
{
    const size_t i = ((size_t)blockIdx.x * 256 + threadIdx.x) * 4;
    if (i >= n) return;
    float4 f = *(const float4*)(in + i);
    uint2 u;
    u.x = bfbits(f.x) | (bfbits(f.y) << 16);
    u.y = bfbits(f.z) | (bfbits(f.w) << 16);
    *(uint2*)(out + i) = u;
}

// ---------------------------------------------------------------------------
// f32 [K][N] -> bf16 [N][K] transpose+convert. 32x32 LDS tiles, 256 threads.
// ---------------------------------------------------------------------------
__global__ __launch_bounds__(256) void cvt_transpose_bf16(
    const float* __restrict__ in, unsigned short* __restrict__ out, int K, int N)
{
    __shared__ float tile[32][33];
    const int n0 = blockIdx.x * 32;
    const int k0 = blockIdx.y * 32;
    const int tx = threadIdx.x & 31;
    const int ty = threadIdx.x >> 5;   // 0..7
#pragma unroll
    for (int i = 0; i < 32; i += 8)
        tile[ty + i][tx] = in[(size_t)(k0 + ty + i) * N + n0 + tx];
    __syncthreads();
#pragma unroll
    for (int i = 0; i < 32; i += 8)
        out[(size_t)(n0 + ty + i) * K + k0 + tx] =
            (unsigned short)bfbits(tile[tx][ty + i]);
}

// ---------------------------------------------------------------------------
// Agent pooling: agent[b,h,p,d] = mean_{j<512} q[b,h,p*512+j,d]
// ---------------------------------------------------------------------------
__global__ __launch_bounds__(64) void pool_kernel(
    const float* __restrict__ qkv, float* __restrict__ agent)
{
    const int bhp = blockIdx.x;
    const int p = bhp % AP;
    const int h = (bhp / AP) % AH;
    const int b = bhp / (AP * AH);
    const int d = threadIdx.x;

    const float* base = qkv + ((size_t)(b * AT + p * TP)) * QKVC + h * AD + d;
    float acc = 0.0f;
    for (int j = 0; j < TP; ++j)
        acc += base[(size_t)j * QKVC];
    agent[(size_t)bhp * AD + d] = acc * (1.0f / (float)TP);
}

// ---------------------------------------------------------------------------
// Stage 1: p1 = softmax_T(agent*SCALE @ k^T);  v_agent = p1 @ v
// One block per (b,h), 256 threads.  Two-pass softmax, exp weights staged in
// global scratch wbuf[bh][p][t]; aggregation over T with 4-way split + LDS.
// ---------------------------------------------------------------------------
__global__ __launch_bounds__(256) void stage1_kernel(
    const float* __restrict__ qkv, const float* __restrict__ agent,
    float* __restrict__ wbuf, float* __restrict__ v_agent)
{
    const int h = blockIdx.x;
    const int b = blockIdx.y;
    const int bh = b * AH + h;
    const int tid = threadIdx.x;

    __shared__ float s_ag[AP][AD];
    __shared__ float red[256 * 8];
    __shared__ float s_m[AP];
    __shared__ float s_rinv[AP];
    __shared__ float parts[4][AP][AD];

    for (int i = tid; i < AP * AD; i += 256)
        s_ag[i >> 6][i & 63] = agent[(size_t)bh * AP * AD + i] * SCALE;
    __syncthreads();

    // ---- pass 1: per-p row max over T ----
    float mx[AP];
#pragma unroll
    for (int p = 0; p < AP; ++p) mx[p] = -1e30f;

    for (int t = tid; t < AT; t += 256) {
        const float* kr = qkv + ((size_t)(b * AT + t)) * QKVC + ADIM + h * AD;
        float dot[AP] = {0.f, 0.f, 0.f, 0.f, 0.f, 0.f, 0.f};
        for (int d = 0; d < AD; ++d) {
            const float kv = kr[d];
#pragma unroll
            for (int p = 0; p < AP; ++p) dot[p] += kv * s_ag[p][d];
        }
#pragma unroll
        for (int p = 0; p < AP; ++p) mx[p] = fmaxf(mx[p], dot[p]);
    }
#pragma unroll
    for (int p = 0; p < AP; ++p) red[tid * 8 + p] = mx[p];
    __syncthreads();
    for (int s = 128; s > 0; s >>= 1) {
        if (tid < s) {
#pragma unroll
            for (int p = 0; p < AP; ++p)
                red[tid * 8 + p] = fmaxf(red[tid * 8 + p], red[(tid + s) * 8 + p]);
        }
        __syncthreads();
    }
    if (tid < AP) s_m[tid] = red[tid];
    __syncthreads();

    // ---- pass 2: exp weights + row sums ----
    float ss[AP] = {0.f, 0.f, 0.f, 0.f, 0.f, 0.f, 0.f};
    for (int t = tid; t < AT; t += 256) {
        const float* kr = qkv + ((size_t)(b * AT + t)) * QKVC + ADIM + h * AD;
        float dot[AP] = {0.f, 0.f, 0.f, 0.f, 0.f, 0.f, 0.f};
        for (int d = 0; d < AD; ++d) {
            const float kv = kr[d];
#pragma unroll
            for (int p = 0; p < AP; ++p) dot[p] += kv * s_ag[p][d];
        }
#pragma unroll
        for (int p = 0; p < AP; ++p) {
            const float e = __expf(dot[p] - s_m[p]);
            wbuf[((size_t)bh * AP + p) * AT + t] = e;
            ss[p] += e;
        }
    }
#pragma unroll
    for (int p = 0; p < AP; ++p) red[tid * 8 + p] = ss[p];
    __syncthreads();
    for (int s = 128; s > 0; s >>= 1) {
        if (tid < s) {
#pragma unroll
            for (int p = 0; p < AP; ++p)
                red[tid * 8 + p] += red[(tid + s) * 8 + p];
        }
        __syncthreads();
    }
    if (tid < AP) s_rinv[tid] = 1.0f / red[tid];
    __threadfence_block();
    __syncthreads();

    // ---- aggregate v_agent[p][d] = (1/sum_p) * sum_t w[p][t] * v[t][d] ----
    const int d = tid & 63;
    const int s = tid >> 6;              // 0..3
    float acc[AP] = {0.f, 0.f, 0.f, 0.f, 0.f, 0.f, 0.f};
    const int t0 = s * (AT / 4);
    for (int t = t0; t < t0 + (AT / 4); ++t) {
        const float vv = qkv[((size_t)(b * AT + t)) * QKVC + 2 * ADIM + h * AD + d];
#pragma unroll
        for (int p = 0; p < AP; ++p)
            acc[p] += wbuf[((size_t)bh * AP + p) * AT + t] * vv;
    }
#pragma unroll
    for (int p = 0; p < AP; ++p) parts[s][p][d] = acc[p];
    __syncthreads();
    if (s == 0) {
#pragma unroll
        for (int p = 0; p < AP; ++p) {
            const float tot = parts[0][p][d] + parts[1][p][d] + parts[2][p][d] + parts[3][p][d];
            v_agent[((size_t)bh * AP + p) * AD + d] = tot * s_rinv[p];
        }
    }
}

// ---------------------------------------------------------------------------
// Stage 2 + depthwise conv + recombine to bf16 [B,T,1024]:
//   p2 = softmax_P(q*SCALE @ agent^T);  out = p2 @ v_agent + dwc(v)
// grid (T/128, H, B), 128 threads; thread owns one t, loops d.
// Output written as bf16 so GEMM-2 consumes it directly.
// ---------------------------------------------------------------------------
__global__ __launch_bounds__(128) void stage2_dwc_kernel(
    const float* __restrict__ qkv, const float* __restrict__ agent,
    const float* __restrict__ v_agent, const float* __restrict__ conv_w,
    const float* __restrict__ conv_b, unsigned short* __restrict__ attn_bf)
{
    const int h = blockIdx.y;
    const int b = blockIdx.z;
    const int bh = b * AH + h;
    const int tid = threadIdx.x;
    const int t = blockIdx.x * 128 + tid;

    __shared__ float s_ag[AP][AD];
    __shared__ float s_va[AP][AD];
    __shared__ float s_cw[AD * 9];
    __shared__ float s_cb[AD];

    for (int i = tid; i < AP * AD; i += 128) {
        s_ag[i >> 6][i & 63] = agent[(size_t)bh * AP * AD + i];
        s_va[i >> 6][i & 63] = v_agent[(size_t)bh * AP * AD + i];
    }
    for (int i = tid; i < AD * 9; i += 128) s_cw[i] = conv_w[i];
    if (tid < AD) s_cb[tid] = conv_b[tid];
    __syncthreads();

    // softmax over P=7 agents
    const float* qr = qkv + ((size_t)(b * AT + t)) * QKVC + h * AD;
    float dot[AP] = {0.f, 0.f, 0.f, 0.f, 0.f, 0.f, 0.f};
    for (int d = 0; d < AD; ++d) {
        const float qv = qr[d] * SCALE;
#pragma unroll
        for (int p = 0; p < AP; ++p) dot[p] += qv * s_ag[p][d];
    }
    float m = dot[0];
#pragma unroll
    for (int p = 1; p < AP; ++p) m = fmaxf(m, dot[p]);
    float w[AP];
    float sum = 0.0f;
#pragma unroll
    for (int p = 0; p < AP; ++p) { w[p] = __expf(dot[p] - m); sum += w[p]; }
    const float rinv = 1.0f / sum;
#pragma unroll
    for (int p = 0; p < AP; ++p) w[p] *= rinv;

    unsigned short* orow = attn_bf + ((size_t)(b * AT + t)) * (AH * AD) + h * AD;

    for (int d = 0; d < AD; ++d) {
        float a = 0.0f;
#pragma unroll
        for (int p = 0; p < AP; ++p) a += w[p] * s_va[p][d];

        float cacc = s_cb[d];
#pragma unroll
        for (int i = 0; i < 3; ++i) {
            const int hh = h + i - 1;
            if (hh < 0 || hh >= AH) continue;
#pragma unroll
            for (int j = 0; j < 3; ++j) {
                const int tt = t + j - 1;
                if (tt < 0 || tt >= AT) continue;
                cacc += s_cw[d * 9 + i * 3 + j] *
                        qkv[((size_t)(b * AT + tt)) * QKVC + 2 * ADIM + hh * AD + d];
            }
        }
        orow[d] = (unsigned short)bfbits(a + cacc);
    }
}

// ---------------------------------------------------------------------------
// Host-side launch
// Inputs: x[4,3584,1024], W_qkv[1024,3072], W_o[1024,1024], b_o[1024],
//         conv_w[64,1,3,3], conv_b[64].  Output: [4,3584,1024] f32.
// ---------------------------------------------------------------------------
extern "C" void kernel_launch(void* const* d_in, const int* in_sizes, int n_in,
                              void* d_out, int out_size, void* d_ws, size_t ws_size,
                              hipStream_t stream) {
    (void)in_sizes; (void)n_in; (void)out_size; (void)ws_size;

    const float* x      = (const float*)d_in[0];
    const float* W_qkv  = (const float*)d_in[1];
    const float* W_o    = (const float*)d_in[2];
    const float* b_o    = (const float*)d_in[3];
    const float* conv_w = (const float*)d_in[4];
    const float* conv_b = (const float*)d_in[5];
    float* out          = (float*)d_out;

    const int M = AB * AT;   // 14336

    // workspace carve-up (bytes; every region 16B-aligned)
    char* w = (char*)d_ws;
    float* qkv = (float*)w;             w += (size_t)M * QKVC * 4;          // 176.2 MB
    float* agent = (float*)w;           w += (size_t)AB * AH * AP * AD * 4; // 114.7 KB
    float* v_agent = (float*)w;         w += (size_t)AB * AH * AP * AD * 4;
    float* wbuf = (float*)w;            w += (size_t)AB * AH * AP * AT * 4; // 6.4 MB
    unsigned short* attn_bf = (unsigned short*)w; w += (size_t)M * ADIM * 2;  // 29.4 MB
    unsigned short* xb = (unsigned short*)w;      w += (size_t)M * ADIM * 2;  // 29.4 MB
    unsigned short* wqkvT = (unsigned short*)w;   w += (size_t)QKVC * ADIM * 2; // 6.3 MB
    unsigned short* woT = (unsigned short*)w;     w += (size_t)ADIM * ADIM * 2; // 2.1 MB

    // 0) precision prep: x -> bf16 rows; weights -> bf16 [N][K] panels
    cvt_f32_bf16<<<((size_t)M * ADIM) / 1024, 256, 0, stream>>>(x, xb, (size_t)M * ADIM);
    cvt_transpose_bf16<<<dim3(QKVC / 32, ADIM / 32), 256, 0, stream>>>(W_qkv, wqkvT, ADIM, QKVC);
    cvt_transpose_bf16<<<dim3(ADIM / 32, ADIM / 32), 256, 0, stream>>>(W_o, woT, ADIM, ADIM);

    // 1) QKV projection (WMMA GEMM, bf16 operands, async LDS staging)
    gemm_bf16_wmma<<<dim3(QKVC / GEMM_BN, M / GEMM_BM), 256, 0, stream>>>(
        xb, wqkvT, nullptr, qkv, M, QKVC, ADIM);

    // 2) agent tokens (adaptive avg pool over T)
    pool_kernel<<<AB * AH * AP, 64, 0, stream>>>(qkv, agent);

    // 3) stage 1: agents attend over T, aggregate v
    stage1_kernel<<<dim3(AH, AB), 256, 0, stream>>>(qkv, agent, wbuf, v_agent);

    // 4) stage 2 softmax over agents + depthwise conv + recombine (bf16 out)
    stage2_dwc_kernel<<<dim3(AT / 128, AH, AB), 128, 0, stream>>>(
        qkv, agent, v_agent, conv_w, conv_b, attn_bf);

    // 5) output projection with bias (WMMA GEMM)
    gemm_bf16_wmma<<<dim3((AH * AD) / GEMM_BN, M / GEMM_BM), 256, 0, stream>>>(
        attn_bf, woT, b_o, out, M, AH * AD, ADIM);
}
